// MambaTrendEncoder_3719441678941
// MI455X (gfx1250) — matrix operand order repaired and checked by reading, not compiled, not run on hardware
//
#include <hip/hip_runtime.h>
#include <math.h>

// ---------------------------------------------------------------------------
// MambaTrendEncoder for MI455X (gfx1250, wave32).
// GEMMs use v_wmma_f32_16x16x32_f16 with fragment-major operand layouts:
//  - B pre-packed so each lane's 16 halves are one aligned 32B vector load
//  - A staged into LDS in fragment order, then hoisted into registers once
//    per wave, so the steady-state N-tile loop is pure B-stream + WMMA.
//  - MODE/REV are template params so store/stage paths are branch-free.
// Selective scan: one wave per channel, lane = state (DS=32), shfl_xor tree.
// ---------------------------------------------------------------------------

#define LTOT 12544   // T*H*W = 4*56*56
#define DIMC 96
#define NLAY 3
#define DIN  192
#define DSN  32
#define DCN  4

typedef _Float16 v16h __attribute__((ext_vector_type(16)));
typedef float    v8f  __attribute__((ext_vector_type(8)));

__device__ __forceinline__ float gelu_exact(float x) {
  return 0.5f * x * (1.0f + erff(x * 0.70710678118654752f));
}
__device__ __forceinline__ float silu_f(float x) {
  return x / (1.0f + expf(-x));
}
__device__ __forceinline__ float softplus_f(float x) {
  return fmaxf(x, 0.0f) + log1pf(expf(-fabsf(x)));
}

// ---- pack fp32 [N][K] weight into WMMA-B fragment order (f16) --------------
// O[((ks*ntiles + nt)*32 + lane)*16 + i] = W[n][k] with
//   n = nt*16 + (lane&15), k = ks*32 + i + 16*(lane>>4)   (zero if n >= N)
__global__ void k_pack_b(const float* __restrict__ W, _Float16* __restrict__ O,
                         int N, int K, int Npad) {
  int o = blockIdx.x * blockDim.x + threadIdx.x;
  if (o >= K * Npad) return;
  int ntiles = Npad >> 4;
  int ks = o / (ntiles << 9);
  int rem = o - ks * (ntiles << 9);
  int nt = rem >> 9;
  int r2 = rem & 511;
  int lane = r2 >> 4;
  int i = r2 & 15;
  int hi = lane >> 4;
  int n = (nt << 4) + (lane & 15);
  int k = (ks << 5) + i + (hi << 4);
  O[o] = (n < N) ? (_Float16)W[n * K + k] : (_Float16)0.0f;
}

// ---- direct 3x3x3 conv, optional instance-norm, exact GELU -----------------
__global__ void k_conv3d(const float* __restrict__ in, const float* __restrict__ w,
                         const float* __restrict__ bias, float* __restrict__ raw,
                         float* __restrict__ out, int Cin, int doNorm) {
  const int c = blockIdx.x;
  const int tid = threadIdx.x;
  __shared__ float rsum[256], rsq[256];
  float lsum = 0.f, lsq = 0.f;
  const float* wc = w + (size_t)c * Cin * 27;
  const float bc = bias[c];
  for (int p = tid; p < LTOT; p += blockDim.x) {
    int t = p / 3136; int r = p - t * 3136; int h = r / 56; int ww = r - h * 56;
    float acc = bc;
    for (int ci = 0; ci < Cin; ++ci) {
      const float* inp = in + (size_t)ci * LTOT;
      const float* wci = wc + ci * 27;
      for (int kt = 0; kt < 3; ++kt) {
        int it = t + kt - 1; if (it < 0 || it >= 4) continue;
        for (int kh = 0; kh < 3; ++kh) {
          int ih = h + kh - 1; if (ih < 0 || ih >= 56) continue;
          for (int kw = 0; kw < 3; ++kw) {
            int iw = ww + kw - 1; if (iw < 0 || iw >= 56) continue;
            acc += wci[kt * 9 + kh * 3 + kw] * inp[it * 3136 + ih * 56 + iw];
          }
        }
      }
    }
    raw[(size_t)c * LTOT + p] = acc;
    lsum += acc; lsq += acc * acc;
  }
  rsum[tid] = lsum; rsq[tid] = lsq;
  __syncthreads();
  for (int st = 128; st > 0; st >>= 1) {
    if (tid < st) { rsum[tid] += rsum[tid + st]; rsq[tid] += rsq[tid + st]; }
    __syncthreads();
  }
  float mean = rsum[0] * (1.0f / LTOT);
  float var  = rsq[0] * (1.0f / LTOT) - mean * mean;
  float rstd = rsqrtf(var + 1e-5f);
  for (int p = tid; p < LTOT; p += blockDim.x) {
    float v = raw[(size_t)c * LTOT + p];
    if (doNorm) v = (v - mean) * rstd;
    out[(size_t)c * LTOT + p] = gelu_exact(v);
  }
}

// ---- antialiased (triangle) resize of pos-embed + transpose add ------------
__global__ void k_pe_add_transpose(const float* __restrict__ f3,
                                   const float* __restrict__ pe,
                                   float* __restrict__ s) {
  int idx = blockIdx.x * blockDim.x + threadIdx.x;
  if (idx >= LTOT * DIMC) return;
  int l = idx / DIMC, c = idx - l * DIMC;
  int t = l / 3136; int r = l - t * 3136; int h = r / 56; int w = r - h * 56;
  const float scl = 56.0f / 120.0f;
  const float inv = 120.0f / 56.0f;
  float ch = (h + 0.5f) * inv - 0.5f;
  float cw = (w + 0.5f) * inv - 0.5f;
  int h0 = (int)ceilf(ch - inv);  if (h0 < 0) h0 = 0;
  int h1 = (int)floorf(ch + inv); if (h1 > 119) h1 = 119;
  int w0 = (int)ceilf(cw - inv);  if (w0 < 0) w0 = 0;
  int w1 = (int)floorf(cw + inv); if (w1 > 119) w1 = 119;
  float hs = 0.f, wsm = 0.f, num = 0.f;
  for (int i = h0; i <= h1; ++i) hs  += fmaxf(0.f, 1.f - fabsf(i - ch) * scl);
  for (int j = w0; j <= w1; ++j) wsm += fmaxf(0.f, 1.f - fabsf(j - cw) * scl);
  const float* base = pe + ((size_t)c * 12 + t) * 14400;
  for (int i = h0; i <= h1; ++i) {
    float wh = fmaxf(0.f, 1.f - fabsf(i - ch) * scl);
    for (int j = w0; j <= w1; ++j) {
      float wwj = fmaxf(0.f, 1.f - fabsf(j - cw) * scl);
      num += wh * wwj * base[i * 120 + j];
    }
  }
  s[idx] = f3[(size_t)c * LTOT + l] + num / (hs * wsm);
}

// ---- LayerNorm over C=96: one wave32 per token, 3 channels per lane --------
__global__ void k_layernorm(const float* __restrict__ in, const float* __restrict__ w,
                            const float* __restrict__ b, float* __restrict__ out) {
  int token = blockIdx.x * (blockDim.x >> 5) + (threadIdx.x >> 5);
  int lane = threadIdx.x & 31;
  if (token >= LTOT) return;
  const float* row = in + token * DIMC;
  float x0 = row[lane], x1 = row[lane + 32], x2 = row[lane + 64];
  float sum = x0 + x1 + x2;
  for (int off = 16; off; off >>= 1) sum += __shfl_xor(sum, off, 32);
  float mean = sum * (1.0f / 96.0f);
  float d0 = x0 - mean, d1 = x1 - mean, d2 = x2 - mean;
  float v = d0 * d0 + d1 * d1 + d2 * d2;
  for (int off = 16; off; off >>= 1) v += __shfl_xor(v, off, 32);
  float rstd = rsqrtf(v * (1.0f / 96.0f) + 1e-5f);
  float* o = out + token * DIMC;
  o[lane]      = d0 * rstd * w[lane]      + b[lane];
  o[lane + 32] = d1 * rstd * w[lane + 32] + b[lane + 32];
  o[lane + 64] = d2 * rstd * w[lane + 64] + b[lane + 64];
}

// ---- WMMA GEMM: D[M x Npad] = A[M x KST*32] (f32) x B (fragment-major) -----
// REV=1 reads A rows reversed (bwd direction). MODE=0: D[row*Npad+col]=acc.
// MODE=1: D[col*M+row] = acc + bias[col] (fused transpose+bias for Wo).
template <int KST, int MODE, int REV>
__global__ void k_gemm_wmma(const float* __restrict__ A, int M,
                            const _Float16* __restrict__ B, int Npad,
                            float* __restrict__ D,
                            const float* __restrict__ bias) {
  constexpr int K = KST * 32;
  alignas(32) __shared__ _Float16 Atile[KST << 9];  // KST x 32 lanes x 16 halves
  const int mtile = blockIdx.x;
  const int tid = threadIdx.x;
  // cooperative A stage: interleave into fragment order at store time.
  for (int o = tid; o < (KST << 9); o += blockDim.x) {
    int ks = o >> 9;
    int rem = o & 511;
    int ln = rem >> 4;
    int i = rem & 15;
    int hh = ln >> 4;
    int row = ln & 15;
    int ka = i + (hh << 4);                             // contiguous k = 16*hh + i (matches B pack)
    int arow = mtile * 16 + row;
    if (REV) arow = M - 1 - arow;
    Atile[o] = (_Float16)A[arow * K + (ks << 5) + ka];
  }
  __syncthreads();
  const int lane = tid & 31;
  const int wv = tid >> 5;
  const int nw = blockDim.x >> 5;
  const int ntiles = Npad >> 4;
  const int hi = lane >> 4;
  const int l15 = lane & 15;
  // hoist all A fragments for this wave into registers (<= 48 VGPRs).
  v16h af[KST];
#pragma unroll
  for (int ks = 0; ks < KST; ++ks)
    af[ks] = *(const v16h*)&Atile[(ks << 9) + (lane << 4)];
  for (int nt = wv; nt < ntiles; nt += nw) {
    // branch-free prefetch of the next N-tile this wave will process.
    int ntp = nt + nw;
    ntp = (ntp < ntiles) ? ntp : nt;                    // v_cndmask, no branch
    __builtin_prefetch(&B[(ntp << 9) + (lane << 4)], 0, 3);
    v8f acc0 = {}, acc1 = {};
#pragma unroll
    for (int ks = 0; ks < KST; ++ks) {
      v16h b = *(const v16h*)&B[((ks * ntiles + nt) << 9) + (lane << 4)];
      if (ks & 1)
        acc1 = __builtin_amdgcn_wmma_f32_16x16x32_f16(
            false, af[ks], false, b, (short)0, acc1, false, false);
      else
        acc0 = __builtin_amdgcn_wmma_f32_16x16x32_f16(
            false, af[ks], false, b, (short)0, acc0, false, false);
    }
    v8f acc = acc0 + acc1;
    const int col = (nt << 4) + l15;
    float bv = 0.0f;
    if (MODE == 1) bv = bias[col];
#pragma unroll
    for (int i = 0; i < 8; ++i) {
      int row = mtile * 16 + i + (hi << 3);   // C/D: VGPR i -> M = i + 8*hi
      if (MODE == 0) D[row * Npad + col] = acc[i];
      else           D[col * M + row] = acc[i] + bv;
    }
  }
}

// ---- depthwise causal conv1d (DC=4, left pad 3) + SiLU ---------------------
__global__ void k_dwconv_silu(const float* __restrict__ xz, const float* __restrict__ cw,
                              const float* __restrict__ cb, float* __restrict__ u) {
  int idx = blockIdx.x * blockDim.x + threadIdx.x;
  if (idx >= LTOT * DIN) return;
  int l = idx / DIN, d = idx - l * DIN;
  float acc = cb[d];
#pragma unroll
  for (int k = 0; k < DCN; ++k) {
    int ll = l - (DCN - 1) + k;
    if (ll >= 0) acc += cw[d * DCN + k] * xz[ll * 384 + d];
  }
  u[idx] = silu_f(acc);
}

// ---- selective scan: one wave per channel d, lane = state s (DS=32) --------
__global__ void k_scan(const float* __restrict__ dbl, const float* __restrict__ u,
                       const float* __restrict__ xz, const float* __restrict__ Wdt,
                       const float* __restrict__ bdt, const float* __restrict__ Alog,
                       const float* __restrict__ Dp, float* __restrict__ yg) {
  int d = blockIdx.x * (blockDim.x >> 5) + (threadIdx.x >> 5);
  int sl = threadIdx.x & 31;
  if (d >= DIN) return;
  float Ads = -expf(Alog[d * DSN + sl]);
  float w0 = Wdt[d * 6 + 0], w1 = Wdt[d * 6 + 1], w2 = Wdt[d * 6 + 2];
  float w3 = Wdt[d * 6 + 3], w4 = Wdt[d * 6 + 4], w5 = Wdt[d * 6 + 5];
  float bd = bdt[d], Dpd = Dp[d];
  float h = 0.f;
  for (int l = 0; l < LTOT; ++l) {
    const float* dr = dbl + l * 80;
    float dtr = bd + dr[0] * w0 + dr[1] * w1 + dr[2] * w2 +
                dr[3] * w3 + dr[4] * w4 + dr[5] * w5;
    float dt = softplus_f(dtr);
    float ul = u[l * DIN + d];
    float Bv = dr[6 + sl], Cv = dr[38 + sl];
    h = expf(dt * Ads) * h + (dt * ul) * Bv;
    float p = h * Cv;
    p += __shfl_xor(p, 16, 32);
    p += __shfl_xor(p, 8, 32);
    p += __shfl_xor(p, 4, 32);
    p += __shfl_xor(p, 2, 32);
    p += __shfl_xor(p, 1, 32);
    if (sl == 0) {
      float z = xz[l * 384 + DIN + d];
      yg[l * DIN + d] = (p + ul * Dpd) * silu_f(z);
    }
  }
}

// ---- residual: s[l] += y_fwd[l] + y_bwd[L-1-l] -----------------------------
__global__ void k_residual(float* __restrict__ s, const float* __restrict__ yf,
                           const float* __restrict__ yb) {
  int idx = blockIdx.x * blockDim.x + threadIdx.x;
  if (idx >= LTOT * DIMC) return;
  int l = idx / DIMC, c = idx - l * DIMC;
  s[idx] += yf[idx] + yb[(LTOT - 1 - l) * DIMC + c];
}

// ---------------------------------------------------------------------------
extern "C" void kernel_launch(void* const* d_in, const int* in_sizes, int n_in,
                              void* d_out, int out_size, void* d_ws, size_t ws_size,
                              hipStream_t stream) {
  const float* x       = (const float*)d_in[0];
  const float* conv1_w = (const float*)d_in[1];
  const float* conv1_b = (const float*)d_in[2];
  const float* conv2_w = (const float*)d_in[3];
  const float* conv2_b = (const float*)d_in[4];
  const float* conv3_w = (const float*)d_in[5];
  const float* conv3_b = (const float*)d_in[6];
  const float* pos     = (const float*)d_in[7];
  const float* ln_w    = (const float*)d_in[8];
  const float* ln_b    = (const float*)d_in[9];
  const float* W_in    = (const float*)d_in[10];
  const float* conv_w  = (const float*)d_in[11];
  const float* conv_b  = (const float*)d_in[12];
  const float* W_x     = (const float*)d_in[13];
  const float* W_dt    = (const float*)d_in[14];
  const float* b_dt    = (const float*)d_in[15];
  const float* A_log   = (const float*)d_in[16];
  const float* Dp      = (const float*)d_in[17];
  const float* W_out   = (const float*)d_in[18];
  const float* lnf_w   = (const float*)d_in[19];
  const float* lnf_b   = (const float*)d_in[20];
  const float* Wo      = (const float*)d_in[21];
  const float* bo      = (const float*)d_in[22];
  float* out = (float*)d_out;

  char* ws = (char*)d_ws;
  size_t off = 0;
  auto take = [&](size_t bytes) -> char* {
    char* p = ws + off;
    off += (bytes + 255) & ~(size_t)255;
    return p;
  };
  float* s_buf   = (float*)take((size_t)LTOT * 96 * 4);
  float* xn_buf  = (float*)take((size_t)LTOT * 96 * 4);
  float* yo0     = (float*)take((size_t)LTOT * 96 * 4);
  float* yo1     = (float*)take((size_t)LTOT * 96 * 4);
  float* u_buf   = (float*)take((size_t)LTOT * 192 * 4);
  float* yg_buf  = (float*)take((size_t)LTOT * 192 * 4);
  float* dbl_buf = (float*)take((size_t)LTOT * 80 * 4);
  float* xz_buf  = (float*)take((size_t)LTOT * 384 * 4);
  _Float16* w_in_h  = (_Float16*)take((size_t)6 * 96 * 384 * 2);
  _Float16* w_x_h   = (_Float16*)take((size_t)6 * 192 * 80 * 2);
  _Float16* w_out_h = (_Float16*)take((size_t)6 * 192 * 96 * 2);
  _Float16* wo_h    = (_Float16*)take((size_t)96 * 96 * 2);
  // stem buffers alias later mamba buffers (stem finishes before layer 0)
  float* f1   = u_buf;    // 24 x L  <= 192 x L
  float* f2   = yg_buf;   // 48 x L  <= 192 x L
  float* f3   = xz_buf;   // 96 x L  <= 384 x L
  float* rawb = dbl_buf;  // 48 x L  <=  80 x L

  // ---- f16 weight packing (fragment-major B) ----
  for (int i = 0; i < 6; ++i) {
    k_pack_b<<<(96 * 384 + 255) / 256, 256, 0, stream>>>(
        W_in + (size_t)i * 384 * 96, w_in_h + (size_t)i * 96 * 384, 384, 96, 384);
    k_pack_b<<<(192 * 80 + 255) / 256, 256, 0, stream>>>(
        W_x + (size_t)i * 70 * 192, w_x_h + (size_t)i * 192 * 80, 70, 192, 80);
    k_pack_b<<<(192 * 96 + 255) / 256, 256, 0, stream>>>(
        W_out + (size_t)i * 96 * 192, w_out_h + (size_t)i * 192 * 96, 96, 192, 96);
  }
  k_pack_b<<<(96 * 96 + 255) / 256, 256, 0, stream>>>(Wo, wo_h, 96, 96, 96);

  // ---- conv stem ----
  k_conv3d<<<24, 256, 0, stream>>>(x,  conv1_w, conv1_b, rawb, f1, 1, 1);
  k_conv3d<<<48, 256, 0, stream>>>(f1, conv2_w, conv2_b, rawb, f2, 24, 1);
  k_conv3d<<<96, 256, 0, stream>>>(f2, conv3_w, conv3_b, f3,   f3, 48, 0);
  k_pe_add_transpose<<<(LTOT * 96 + 255) / 256, 256, 0, stream>>>(f3, pos, s_buf);

  // ---- mamba layers ----
  const int gM = LTOT / 16;  // 784 M-tiles
  for (int i = 0; i < NLAY; ++i) {
    k_layernorm<<<LTOT / 4, 128, 0, stream>>>(s_buf, ln_w + i * 96, ln_b + i * 96, xn_buf);
    for (int dir = 0; dir < 2; ++dir) {
      int sl = i * 2 + dir;
      if (dir == 0)
        k_gemm_wmma<3, 0, 0><<<gM, 128, 0, stream>>>(xn_buf, LTOT,
            w_in_h + (size_t)sl * 96 * 384, 384, xz_buf, nullptr);
      else
        k_gemm_wmma<3, 0, 1><<<gM, 128, 0, stream>>>(xn_buf, LTOT,
            w_in_h + (size_t)sl * 96 * 384, 384, xz_buf, nullptr);
      k_dwconv_silu<<<(LTOT * 192 + 255) / 256, 256, 0, stream>>>(
          xz_buf, conv_w + (size_t)sl * 192 * 4, conv_b + (size_t)sl * 192, u_buf);
      k_gemm_wmma<6, 0, 0><<<gM, 128, 0, stream>>>(u_buf, LTOT,
          w_x_h + (size_t)sl * 192 * 80, 80, dbl_buf, nullptr);
      k_scan<<<48, 128, 0, stream>>>(dbl_buf, u_buf, xz_buf,
          W_dt + (size_t)sl * 192 * 6, b_dt + (size_t)sl * 192,
          A_log + (size_t)sl * 192 * 32, Dp + (size_t)sl * 192, yg_buf);
      k_gemm_wmma<6, 0, 0><<<gM, 128, 0, stream>>>(yg_buf, LTOT,
          w_out_h + (size_t)sl * 192 * 96, 96, dir ? yo1 : yo0, nullptr);
    }
    k_residual<<<(LTOT * 96 + 255) / 256, 256, 0, stream>>>(s_buf, yo0, yo1);
  }

  // ---- final LN + projection, fused transpose to (96, T, H, W) ----
  k_layernorm<<<LTOT / 4, 128, 0, stream>>>(s_buf, lnf_w, lnf_b, xn_buf);
  k_gemm_wmma<3, 1, 0><<<gM, 128, 0, stream>>>(xn_buf, LTOT, wo_h, 96, out, bo);

  (void)in_sizes; (void)n_in; (void)out_size; (void)ws_size;
}
